// SelfAttentionWithRoPE_59193239273526
// MI455X (gfx1250) — matrix operand order repaired and checked by reading, not compiled, not run on hardware
//
#include <hip/hip_runtime.h>

// ---------------------------------------------------------------------------
// SelfAttentionWithRoPE for MI455X (gfx1250, wave32, WMMA, TDM).
// Pipeline: f32->bf16 converts -> QKV GEMM (bf16 WMMA, f32 acc) -> RoPE ->
// V transpose -> per-(head,16-query-tile) flash-style attention (TDM-staged
// K tiles, writes probs + bf16 attn) -> out-projection GEMM.
// HBM-write-bound on the 553MB probs -> all probs stores are b128.
// ---------------------------------------------------------------------------

typedef __bf16 bf16_t;
typedef __attribute__((ext_vector_type(8)))  bf16_t v8bf;
typedef __attribute__((ext_vector_type(16))) bf16_t v16bf;
typedef __attribute__((ext_vector_type(8)))  float  v8f;
typedef __attribute__((ext_vector_type(4)))  unsigned int v4u;
typedef __attribute__((ext_vector_type(8)))  int v8i;
typedef __attribute__((ext_vector_type(4)))  int v4i;

#define SEQ   2048
#define EMB   1024
#define NH    16
#define HD    64
#define NEG_INF (-__builtin_inff())

static __device__ __forceinline__ v8f vzero8() {
  v8f z = {0.f, 0.f, 0.f, 0.f, 0.f, 0.f, 0.f, 0.f};
  return z;
}

static __device__ __forceinline__ v16bf combine8(v8bf lo, v8bf hi) {
  return __builtin_shufflevector(lo, hi, 0, 1, 2, 3, 4, 5, 6, 7,
                                         8, 9, 10, 11, 12, 13, 14, 15);
}

static __device__ __forceinline__ v8f wmma_bf16(v16bf a, v16bf b, v8f c) {
  // D = A(16x32 bf16) * B(32x16 bf16) + C(16x16 f32)
  return __builtin_amdgcn_wmma_f32_16x16x32_bf16(
      /*neg_a=*/false, a, /*neg_b=*/false, b,
      /*c_mod=*/(short)0, c, /*reuse_a=*/false, /*reuse_b=*/false);
}

// A-fragment (16x32 bf16) from row-major: lane L: M=L%16,
// K = hi*16 .. hi*16+15 contiguous, hi = L/16.
static __device__ __forceinline__ v16bf load_afrag(const bf16_t* row_base,
                                                   int hi) {
  const bf16_t* p = row_base + hi * 16;
  return combine8(*(const v8bf*)p, *(const v8bf*)(p + 8));
}

// ---------------------------------------------------------------------------
// Tensor Data Mover: async-load a 32(keys) x 64(head-dim) bf16 tile whose
// rows are EMB elements apart into LDS (packed 32x64). D# per ISA 08 §8.
// This toolchain's builtin takes 6 args (g0, g1, g2, g3, extra, cpol).
// ---------------------------------------------------------------------------
static __device__ __forceinline__ void tdm_load_k_chunk(const bf16_t* gsrc,
                                                        bf16_t* ldst) {
  // Low 32 bits of a generic LDS pointer are the LDS byte offset
  // (aperture occupies bits 63:32 per CDNA5 ISA).
  unsigned lds_addr = (unsigned)(size_t)ldst;
  unsigned long long ga = (unsigned long long)(size_t)gsrc;

  v4u g0;
  g0[0] = 1u;                                   // count=1, user descriptor
  g0[1] = lds_addr;                             // lds_addr [63:32]
  g0[2] = (unsigned)ga;                         // global_addr lo
  g0[3] = (unsigned)(ga >> 32) | 0x80000000u;   // global_addr hi | type=2

  v8i g1;
  g1[0] = 0x00010000;         // workgroup_mask=0, data_size=1 (2 bytes)
  g1[1] = (int)(64u << 16);   // tensor_dim0 = 64  (bits 79:48)
  g1[2] = (int)(32u << 16);   // tensor_dim1 = 32  (bits 111:80)
  g1[3] = (int)(64u << 16);   // tile_dim0  = 64  (bits 127:112)
  g1[4] = 32;                 // tile_dim1  = 32
  g1[5] = EMB;                // tensor_dim0_stride = 1024 (elements)
  g1[6] = 0;
  g1[7] = 0;

  v4i z4 = {0, 0, 0, 0};      // 2D tensor: groups 2/3 unused
  v8i z8 = {0, 0, 0, 0, 0, 0, 0, 0};
  __builtin_amdgcn_tensor_load_to_lds(g0, g1, z4, z4, z8, 0);
}

// ---------------------------------------------------------------------------
// f32 -> bf16 convert (grid-stride)
// ---------------------------------------------------------------------------
__global__ __launch_bounds__(256) void f32_to_bf16_kernel(
    const float* __restrict__ in, bf16_t* __restrict__ out, int n) {
  for (int i = blockIdx.x * blockDim.x + threadIdx.x; i < n;
       i += gridDim.x * blockDim.x)
    out[i] = (bf16_t)in[i];
}

// ---------------------------------------------------------------------------
// C[M,N] = A[M,K](bf16) * B[N,K](bf16)^T + bias[N]   (f32 out)
// block = 128 threads = 4 waves; block tile 64(M) x 64(N); wave: 16x64 strip.
// ---------------------------------------------------------------------------
__global__ __launch_bounds__(128) void gemm_bf16_nt_kernel(
    const bf16_t* __restrict__ A, const bf16_t* __restrict__ B,
    const float* __restrict__ bias, float* __restrict__ C, int N, int K) {
  const int lane = threadIdx.x & 31;
  const int wave = threadIdx.x >> 5;
  const int lm = lane & 15;
  const int hi = lane >> 4;
  const int m0 = blockIdx.y * 64 + wave * 16;
  const int n0 = blockIdx.x * 64;

  v8f acc[4];
#pragma unroll
  for (int t = 0; t < 4; ++t) acc[t] = vzero8();

  const bf16_t* arow = A + (size_t)(m0 + lm) * K;
  for (int k0 = 0; k0 < K; k0 += 32) {
    v16bf af = load_afrag(arow + k0, hi);
    __builtin_prefetch(arow + k0 + 64, 0, 1);  // global_prefetch_b8
#pragma unroll
    for (int t = 0; t < 4; ++t) {
      // B-fragment: lane: N = n0+t*16+lm, K = k0 + hi*16 + [0..15] contiguous
      const bf16_t* bp = B + (size_t)(n0 + t * 16 + lm) * K + k0 + hi * 16;
      v16bf bfrag = *(const v16bf*)bp;
      acc[t] = wmma_bf16(af, bfrag, acc[t]);
    }
  }

#pragma unroll
  for (int t = 0; t < 4; ++t) {
    const int col = n0 + t * 16 + lm;
    const float bv = bias[col];
#pragma unroll
    for (int r = 0; r < 8; ++r) {
      const int row = m0 + r + hi * 8;
      C[(size_t)row * N + col] = acc[t][r] + bv;
    }
  }
}

// ---------------------------------------------------------------------------
// RoPE: qkv_f32 [S,3*E] -> qr/kr bf16 [S,E] for both shifts.
// 1/sqrt(D) folded into q. One thread per (s, h, j) with j in [0,32).
// ---------------------------------------------------------------------------
__global__ __launch_bounds__(256) void rope_kernel(
    const float* __restrict__ qkv, bf16_t* __restrict__ qr16,
    bf16_t* __restrict__ kr16, bf16_t* __restrict__ qr32,
    bf16_t* __restrict__ kr32) {
  const int idx = blockIdx.x * blockDim.x + threadIdx.x;  // S*NH*32 total
  const int j = idx & 31;
  const int h = (idx >> 5) & (NH - 1);
  const int s = idx >> 9;
  if (s >= SEQ) return;

  const float inv_freq = __powf(10000.f, -(float)(2 * j) / (float)HD);
  const float qscale = 0.125f;  // 1/sqrt(64)

  const size_t ib = (size_t)s * (3 * EMB) + h * HD + j;
  const float qo = qkv[ib], qe = qkv[ib + 32];
  const float ko = qkv[ib + EMB], ke = qkv[ib + EMB + 32];
  const size_t ob = (size_t)s * EMB + h * HD + j;

  {
    const float ang = (float)(s + 16) * inv_freq;
    const float sn = __sinf(ang), cs = __cosf(ang);
    qr16[ob] = (bf16_t)((qo * cs - qe * sn) * qscale);
    qr16[ob + 32] = (bf16_t)((qe * cs + qo * sn) * qscale);
    kr16[ob] = (bf16_t)(ko * cs - ke * sn);
    kr16[ob + 32] = (bf16_t)(ke * cs + ko * sn);
  }
  {
    const float ang = (float)(s + 32) * inv_freq;
    const float sn = __sinf(ang), cs = __cosf(ang);
    qr32[ob] = (bf16_t)((qo * cs - qe * sn) * qscale);
    qr32[ob + 32] = (bf16_t)((qe * cs + qo * sn) * qscale);
    kr32[ob] = (bf16_t)(ko * cs - ke * sn);
    kr32[ob + 32] = (bf16_t)(ke * cs + ko * sn);
  }
}

// ---------------------------------------------------------------------------
// V transpose: vT[(h*HD+d)*S + s] = bf16(qkv[s, 2*E + h*HD + d])
// ---------------------------------------------------------------------------
__global__ __launch_bounds__(256) void vtrans_kernel(
    const float* __restrict__ qkv, bf16_t* __restrict__ vT) {
  const int idx = blockIdx.x * blockDim.x + threadIdx.x;  // EMB*SEQ total
  if (idx >= EMB * SEQ) return;
  const int row = idx >> 11;  // h*HD + d
  const int s = idx & (SEQ - 1);
  vT[idx] = (bf16_t)qkv[(size_t)s * (3 * EMB) + 2 * EMB + row];
}

// ---------------------------------------------------------------------------
// Attention for one shift. Block = 1 wave (32 threads).
// grid = (SEQ/16 query tiles, NH heads). Two-pass flash softmax.
// Pass 2 K tiles double-buffered into LDS via the Tensor Data Mover.
// Writes probs (f32, causal-zeroed, [h][q][k]) and attn (bf16 [S,E]).
// ---------------------------------------------------------------------------
__global__ __launch_bounds__(32) void attn_kernel(
    const bf16_t* __restrict__ qr, const bf16_t* __restrict__ kr,
    const bf16_t* __restrict__ vT, float* __restrict__ probs,
    bf16_t* __restrict__ attn) {
  __shared__ __align__(16) float pf[16][32];   // probs staging (b128 stores)
  __shared__ __align__(16) bf16_t pb[16][32];  // C/D -> A relayout for P.V
  __shared__ __align__(16) bf16_t kbuf[2][32][64];  // TDM-staged K chunks

  const int lane = threadIdx.x;
  const int lm = lane & 15;
  const int hi = lane >> 4;
  const int qt = blockIdx.x;
  const int h = blockIdx.y;
  const int q0 = qt * 16;

  // Q A-fragments (k-dim = head dim, 64 -> two 16x32 fragments), reused.
  const bf16_t* qrow = qr + (size_t)(q0 + lm) * EMB + h * HD;
  const v16bf aq0 = load_afrag(qrow, hi);
  const v16bf aq1 = load_afrag(qrow + 32, hi);

  // ---- pass 1: online row max / row sum over key tiles 0..qt ----
  float Mr[8], Lr[8];
#pragma unroll
  for (int r = 0; r < 8; ++r) { Mr[r] = NEG_INF; Lr[r] = 0.f; }

  for (int kt = 0; kt <= qt; ++kt) {
    const int k0 = kt * 16;
    const bf16_t* krow = kr + (size_t)(k0 + lm) * EMB + h * HD;
    v16bf b0 = *(const v16bf*)(krow + hi * 16);
    v16bf b1 = *(const v16bf*)(krow + 32 + hi * 16);
    v8f s = vzero8();
    s = wmma_bf16(aq0, b0, s);
    s = wmma_bf16(aq1, b1, s);
    const bool diag = (kt == qt);
#pragma unroll
    for (int r = 0; r < 8; ++r) {
      const int row = r + hi * 8;            // local query row 0..15
      float sv = s[r];
      if (diag && (lm > row)) sv = NEG_INF;  // causal mask
      float tm = sv;
      tm = fmaxf(tm, __shfl_xor(tm, 1));
      tm = fmaxf(tm, __shfl_xor(tm, 2));
      tm = fmaxf(tm, __shfl_xor(tm, 4));
      tm = fmaxf(tm, __shfl_xor(tm, 8));
      const float nM = fmaxf(Mr[r], tm);
      const float resc = (Mr[r] == NEG_INF) ? 0.f : __expf(Mr[r] - nM);
      float e = (sv == NEG_INF) ? 0.f : __expf(sv - nM);
      e += __shfl_xor(e, 1);
      e += __shfl_xor(e, 2);
      e += __shfl_xor(e, 4);
      e += __shfl_xor(e, 8);
      Lr[r] = Lr[r] * resc + e;
      Mr[r] = nM;
    }
  }
  float invL[8];
#pragma unroll
  for (int r = 0; r < 8; ++r) invL[r] = 1.0f / Lr[r];

  // ---- pass 2: normalized probs + P.V, K chunks via TDM double-buffer ----
  const size_t pbase = (size_t)h * SEQ * SEQ;
  v8f acc[4];
#pragma unroll
  for (int t = 0; t < 4; ++t) acc[t] = vzero8();

  const bf16_t* kgbase = kr + h * HD;
  const int nchunks = qt / 2 + 1;  // 32-key chunks covering tiles 0..qt

  tdm_load_k_chunk(kgbase, &kbuf[0][0][0]);  // prefetch chunk 0

  for (int c = 0; c < nchunks; ++c) {
    const int kc0 = c * 32;
    if (c + 1 < nchunks) {  // prefetch next chunk, wait for current
      tdm_load_k_chunk(kgbase + (size_t)(kc0 + 32) * EMB,
                       &kbuf[(c + 1) & 1][0][0]);
      __builtin_amdgcn_s_wait_tensorcnt(1);
    } else {
      __builtin_amdgcn_s_wait_tensorcnt(0);
    }
    asm volatile("" ::: "memory");  // TDM wrote LDS behind the compiler

#pragma unroll
    for (int t = 0; t < 2; ++t) {
      const int kt = 2 * c + t;
      float pv[8];
      if (kt <= qt) {
        const bf16_t* krow = &kbuf[c & 1][t * 16 + lm][0];
        v16bf b0 = *(const v16bf*)(krow + hi * 16);
        v16bf b1 = *(const v16bf*)(krow + 32 + hi * 16);
        v8f s = vzero8();
        s = wmma_bf16(aq0, b0, s);
        s = wmma_bf16(aq1, b1, s);
        const bool diag = (kt == qt);
#pragma unroll
        for (int r = 0; r < 8; ++r) {
          const int row = r + hi * 8;
          const bool masked = diag && (lm > row);
          pv[r] = masked ? 0.f : __expf(s[r] - Mr[r]) * invL[r];
        }
      } else {
#pragma unroll
        for (int r = 0; r < 8; ++r) pv[r] = 0.f;
      }
#pragma unroll
      for (int r = 0; r < 8; ++r) {
        const int row = r + hi * 8;
        pf[row][t * 16 + lm] = pv[r];
        pb[row][t * 16 + lm] = (bf16_t)pv[r];
      }
    }
    __syncthreads();

    // probs stores as b128: 4 rows x 8 lanes x float4 per iteration
#pragma unroll
    for (int mg = 0; mg < 4; ++mg) {
      const int m = mg * 4 + (lane >> 3);
      const int cj = (lane & 7) * 4;
      float4 v = *(const float4*)&pf[m][cj];
      *(float4*)(probs + pbase + (size_t)(q0 + m) * SEQ + kc0 + cj) = v;
    }

    // P.V: A = P chunk (16x32 from LDS), B = vT (k=key contiguous)
    {
      v16bf ap = load_afrag(&pb[lm][0], hi);
#pragma unroll
      for (int t = 0; t < 4; ++t) {
        const bf16_t* bp =
            vT + (size_t)(h * HD + t * 16 + lm) * SEQ + kc0 + hi * 16;
        v16bf bfrag = *(const v16bf*)bp;
        acc[t] = wmma_bf16(ap, bfrag, acc[t]);
      }
    }
    __syncthreads();
  }

  // zero-fill probs beyond causal region (b128 stores after 128-alignment)
  int zk = nchunks * 32;
  for (; (zk & 127) != 0 && zk < SEQ; zk += 32)
#pragma unroll 4
    for (int m = 0; m < 16; ++m)
      probs[pbase + (size_t)(q0 + m) * SEQ + zk + lane] = 0.f;
  const float4 zf4 = make_float4(0.f, 0.f, 0.f, 0.f);
  for (; zk < SEQ; zk += 128)
#pragma unroll 4
    for (int m = 0; m < 16; ++m)
      *(float4*)(probs + pbase + (size_t)(q0 + m) * SEQ + zk + lane * 4) = zf4;

  // attn out (bf16, [S, E] layout for the out-projection GEMM)
#pragma unroll
  for (int t = 0; t < 4; ++t)
#pragma unroll
    for (int r = 0; r < 8; ++r)
      attn[(size_t)(q0 + r + hi * 8) * EMB + h * HD + t * 16 + lm] =
          (bf16_t)acc[t][r];
}

// ---------------------------------------------------------------------------
// launch
// ---------------------------------------------------------------------------
extern "C" void kernel_launch(void* const* d_in, const int* in_sizes, int n_in,
                              void* d_out, int out_size, void* d_ws,
                              size_t ws_size, hipStream_t stream) {
  (void)in_sizes; (void)n_in; (void)out_size; (void)ws_size;

  const float* x = (const float*)d_in[0];      // [S, E]
  const float* w_qkv = (const float*)d_in[1];  // [3E, E]
  const float* b_qkv = (const float*)d_in[2];  // [3E]
  const float* w_out = (const float*)d_in[3];  // [E, E]
  const float* b_out = (const float*)d_in[4];  // [E]
  float* out = (float*)d_out;

  // workspace partition (64 MB total)
  char* ws = (char*)d_ws;
  bf16_t* xbf = (bf16_t*)ws;            ws += (size_t)SEQ * EMB * 2;
  bf16_t* wqkvbf = (bf16_t*)ws;         ws += (size_t)3 * EMB * EMB * 2;
  bf16_t* woutbf = (bf16_t*)ws;         ws += (size_t)EMB * EMB * 2;
  float*  qkvf = (float*)ws;            ws += (size_t)SEQ * 3 * EMB * 4;
  bf16_t* qr16 = (bf16_t*)ws;           ws += (size_t)SEQ * EMB * 2;
  bf16_t* kr16 = (bf16_t*)ws;           ws += (size_t)SEQ * EMB * 2;
  bf16_t* qr32 = (bf16_t*)ws;           ws += (size_t)SEQ * EMB * 2;
  bf16_t* kr32 = (bf16_t*)ws;           ws += (size_t)SEQ * EMB * 2;
  bf16_t* vT = (bf16_t*)ws;             ws += (size_t)EMB * SEQ * 2;
  bf16_t* attn16 = (bf16_t*)ws;         ws += (size_t)SEQ * EMB * 2;
  bf16_t* attn32 = (bf16_t*)ws;         ws += (size_t)SEQ * EMB * 2;

  const size_t P = (size_t)NH * SEQ * SEQ;  // probs elements per shift
  const size_t A = (size_t)SEQ * EMB;       // attn-out elements per shift

  // 1) precision downconverts
  f32_to_bf16_kernel<<<1024, 256, 0, stream>>>(x, xbf, SEQ * EMB);
  f32_to_bf16_kernel<<<1024, 256, 0, stream>>>(w_qkv, wqkvbf, 3 * EMB * EMB);
  f32_to_bf16_kernel<<<1024, 256, 0, stream>>>(w_out, woutbf, EMB * EMB);

  // 2) QKV projection: [S,3E] = x * w_qkv^T + b_qkv
  gemm_bf16_nt_kernel<<<dim3(3 * EMB / 64, SEQ / 64), 128, 0, stream>>>(
      xbf, wqkvbf, b_qkv, qkvf, 3 * EMB, EMB);

  // 3) RoPE (both shifts) + V transpose
  rope_kernel<<<(SEQ * NH * 32) / 256, 256, 0, stream>>>(qkvf, qr16, kr16,
                                                         qr32, kr32);
  vtrans_kernel<<<(EMB * SEQ) / 256, 256, 0, stream>>>(qkvf, vT);

  // 4) attention per shift: probs -> d_out, attn -> bf16 scratch
  attn_kernel<<<dim3(SEQ / 16, NH), 32, 0, stream>>>(qr16, kr16, vT, out,
                                                     attn16);
  attn_kernel<<<dim3(SEQ / 16, NH), 32, 0, stream>>>(qr32, kr32, vT,
                                                     out + P + A, attn32);

  // 5) out-projections
  gemm_bf16_nt_kernel<<<dim3(EMB / 64, SEQ / 64), 128, 0, stream>>>(
      attn16, woutbf, b_out, out + P, EMB, EMB);
  gemm_bf16_nt_kernel<<<dim3(EMB / 64, SEQ / 64), 128, 0, stream>>>(
      attn32, woutbf, b_out, out + 2 * P + A, EMB, EMB);
}
